// GCNEncoder_60593398612124
// MI455X (gfx1250) — matrix operand order repaired and checked by reading, not compiled, not run on hardware
//
#include <hip/hip_runtime.h>
#include <hip/hip_bf16.h>

typedef __attribute__((ext_vector_type(16))) _Float16 v16h;
typedef __attribute__((ext_vector_type(8)))  _Float16 v8h;
typedef __attribute__((ext_vector_type(8)))  float    v8f;

#define IN_C 128
#define HID  64
#define OUT_C 32

// ---------------- degree / norm ----------------

__global__ void init_deg_kernel(float* __restrict__ deg, int n) {
  int i = blockIdx.x * blockDim.x + threadIdx.x;
  if (i < n) deg[i] = 1.0f;  // self-loop contributes 1 to every node's degree
}

__global__ void count_deg_kernel(const long long* __restrict__ ei,
                                 float* __restrict__ deg, long long E) {
  long long e = (long long)blockIdx.x * blockDim.x + threadIdx.x;
  if (e < E) atomicAdd(deg + ei[E + e], 1.0f);
}

__global__ void dinv_kernel(float* __restrict__ deg, int n) {
  int i = blockIdx.x * blockDim.x + threadIdx.x;
  if (i < n) deg[i] = rsqrtf(deg[i]);  // deg >= 1 always (self-loops)
}

// ---------------- weight prep: transpose + f32 -> f16 ----------------
// w1t[n][k] = W1[k][n]  (64 x 128), w2t[n][k] = W2[k][n]  (32 x 64)
__global__ void prep_w_kernel(const float* __restrict__ W1, const float* __restrict__ W2,
                              _Float16* __restrict__ w1t, _Float16* __restrict__ w2t) {
  int t = blockIdx.x * blockDim.x + threadIdx.x;
  if (t < HID * IN_C) {
    int nIdx = t / IN_C, k = t % IN_C;
    w1t[t] = (_Float16)W1[k * HID + nIdx];
  } else if (t < HID * IN_C + OUT_C * HID) {
    int u = t - HID * IN_C;
    int nIdx = u / HID, k = u % HID;
    w2t[u] = (_Float16)W2[k * OUT_C + nIdx];
  }
}

// ---------------- GEMM1: t1 = x @ W1 (f16 out), agg1 = t1 * dinv^2 (f32) ----
// Block = 128 threads = 4 waves; each wave computes a 16(M) x 64(N) tile.
// A (16x32 f16): lane l holds row M=l%16; K = c0 + 0..15 contiguous
//                with c0 = kk*32 + (l/16)*16.
// B (32x16 f16): lane l holds col N=l%16; K = kk*32 + (l/16)*16 + 0..15 contiguous
//                -> one v16h load from the transposed weight matrix.
__global__ __launch_bounds__(128) void gemm1_kernel(
    const float* __restrict__ x, const _Float16* __restrict__ w1t,
    const float* __restrict__ dinv,
    _Float16* __restrict__ t1, float* __restrict__ agg1, int n)
{
  const int lane = threadIdx.x & 31;
  const int lid  = lane & 15;
  const int hi   = lane >> 4;
  const int rowBase = blockIdx.x * 64 + (threadIdx.x >> 5) * 16;
  int m = rowBase + lid;
  if (m >= n) m = n - 1;  // clamp for loads; padding rows never stored

  v8f acc0 = {}, acc1 = {}, acc2 = {}, acc3 = {};
  #pragma unroll
  for (int kk = 0; kk < 4; ++kk) {
    const float* xr = x + (size_t)m * IN_C + kk * 32 + hi * 16;
    v16h a;
    #pragma unroll
    for (int i = 0; i < 8; ++i) {
      a[i]     = (_Float16)xr[i];        // K = c0 + i
      a[8 + i] = (_Float16)xr[8 + i];    // K = c0 + 8 + i
    }
    const _Float16* wb = w1t + lid * IN_C + kk * 32 + hi * 16;
    v16h bf0 = *(const v16h*)(wb + 0  * 16 * IN_C);
    v16h bf1 = *(const v16h*)(wb + 1  * 16 * IN_C);
    v16h bf2 = *(const v16h*)(wb + 2  * 16 * IN_C);
    v16h bf3 = *(const v16h*)(wb + 3  * 16 * IN_C);
    acc0 = __builtin_amdgcn_wmma_f32_16x16x32_f16(false, a, false, bf0, (short)0, acc0, false, false);
    acc1 = __builtin_amdgcn_wmma_f32_16x16x32_f16(false, a, false, bf1, (short)0, acc1, false, false);
    acc2 = __builtin_amdgcn_wmma_f32_16x16x32_f16(false, a, false, bf2, (short)0, acc2, false, false);
    acc3 = __builtin_amdgcn_wmma_f32_16x16x32_f16(false, a, false, bf3, (short)0, acc3, false, false);
  }
  // C layout: VGPR r, lane l -> row = rowBase + (l/16)*8 + r, col = j*16 + l%16
  #pragma unroll
  for (int r = 0; r < 8; ++r) {
    int row = rowBase + hi * 8 + r;
    if (row < n) {
      float dv = dinv[row];
      float dsq = dv * dv;
      size_t base = (size_t)row * HID + lid;
      float v0 = acc0[r], v1 = acc1[r], v2 = acc2[r], v3 = acc3[r];
      t1[base +  0] = (_Float16)v0;  agg1[base +  0] = v0 * dsq;  // self-loop init
      t1[base + 16] = (_Float16)v1;  agg1[base + 16] = v1 * dsq;
      t1[base + 32] = (_Float16)v2;  agg1[base + 32] = v2 * dsq;
      t1[base + 48] = (_Float16)v3;  agg1[base + 48] = v3 * dsq;
    }
  }
}

// ---------------- edge scatter: acc[dst] += feat[src] * dinv[src]*dinv[dst] ---
template <int C, int CLOG2>
__global__ void scatter_kernel(const long long* __restrict__ ei,
                               const _Float16* __restrict__ feat,
                               const float* __restrict__ dinv,
                               float* __restrict__ acc, long long E)
{
  long long t = (long long)blockIdx.x * blockDim.x + threadIdx.x;
  if (t >= (E << CLOG2)) return;
  long long e = t >> CLOG2;
  int c = (int)(t & (C - 1));
  long long s = ei[e];
  long long d = ei[E + e];
  float nrm = dinv[s] * dinv[d];
  atomicAdd(acc + d * C + c, (float)feat[s * C + c] * nrm);
}

// ---------------- bias + ReLU, f32 -> f16 for second gather -----------------
__global__ void relu_bias_kernel(const float* __restrict__ agg1,
                                 const float* __restrict__ b1,
                                 _Float16* __restrict__ h, long long total)
{
  long long t = (long long)blockIdx.x * blockDim.x + threadIdx.x;
  if (t < total) {
    int c = (int)(t & (HID - 1));
    float v = agg1[t] + b1[c];
    h[t] = (_Float16)(v > 0.0f ? v : 0.0f);
  }
}

// ---------------- GEMM2: t2 = h @ W2 (f16), out init = t2*dinv^2 + b2 -------
__global__ __launch_bounds__(128) void gemm2_kernel(
    const _Float16* __restrict__ h, const _Float16* __restrict__ w2t,
    const float* __restrict__ dinv, const float* __restrict__ b2,
    _Float16* __restrict__ t2, float* __restrict__ out, int n)
{
  const int lane = threadIdx.x & 31;
  const int lid  = lane & 15;
  const int hi   = lane >> 4;
  const int rowBase = blockIdx.x * 64 + (threadIdx.x >> 5) * 16;
  int m = rowBase + lid;
  if (m >= n) m = n - 1;

  v8f acc0 = {}, acc1 = {};
  #pragma unroll
  for (int kk = 0; kk < 2; ++kk) {
    const _Float16* hr = h + (size_t)m * HID + kk * 32 + hi * 16;
    v8h lo  = *(const v8h*)(hr);
    v8h hi8 = *(const v8h*)(hr + 8);
    v16h a;
    #pragma unroll
    for (int i = 0; i < 8; ++i) { a[i] = lo[i]; a[8 + i] = hi8[i]; }
    const _Float16* wb = w2t + lid * HID + kk * 32 + hi * 16;
    v16h bf0 = *(const v16h*)(wb);
    v16h bf1 = *(const v16h*)(wb + 16 * HID);
    acc0 = __builtin_amdgcn_wmma_f32_16x16x32_f16(false, a, false, bf0, (short)0, acc0, false, false);
    acc1 = __builtin_amdgcn_wmma_f32_16x16x32_f16(false, a, false, bf1, (short)0, acc1, false, false);
  }
  #pragma unroll
  for (int r = 0; r < 8; ++r) {
    int row = rowBase + hi * 8 + r;
    if (row < n) {
      float dv = dinv[row];
      float dsq = dv * dv;
      size_t base = (size_t)row * OUT_C + lid;
      float v0 = acc0[r], v1 = acc1[r];
      t2[base +  0] = (_Float16)v0;  out[base +  0] = v0 * dsq + b2[lid];
      t2[base + 16] = (_Float16)v1;  out[base + 16] = v1 * dsq + b2[lid + 16];
    }
  }
}

// ---------------- host launch ----------------

extern "C" void kernel_launch(void* const* d_in, const int* in_sizes, int n_in,
                              void* d_out, int out_size, void* d_ws, size_t ws_size,
                              hipStream_t stream) {
  (void)n_in; (void)out_size; (void)ws_size;
  const float*     x  = (const float*)d_in[0];
  const long long* ei = (const long long*)d_in[1];   // edge_index is int64
  const float*     W1 = (const float*)d_in[2];
  const float*     b1 = (const float*)d_in[3];
  const float*     W2 = (const float*)d_in[4];
  const float*     b2 = (const float*)d_in[5];
  float* out = (float*)d_out;

  const int n = in_sizes[0] / IN_C;
  const long long E = in_sizes[1] / 2;

  char* ws = (char*)d_ws;
  size_t off = 0;
  auto take = [&](size_t bytes) -> char* {
    char* p = ws + off;
    off = (off + bytes + 255) & ~(size_t)255;
    return p;
  };
  float*     dinv = (float*)take((size_t)n * sizeof(float));           // deg -> dinv in place
  _Float16*  w1t  = (_Float16*)take((size_t)HID * IN_C * 2);
  _Float16*  w2t  = (_Float16*)take((size_t)OUT_C * HID * 2);
  _Float16*  t1   = (_Float16*)take((size_t)n * HID * 2);              // x@W1 f16; later h f16
  float*     agg1 = (float*)take((size_t)n * HID * sizeof(float));     // fp32 scatter target
  _Float16*  t2   = (_Float16*)take((size_t)n * OUT_C * 2);            // h@W2 f16

  const dim3 b256(256);
  init_deg_kernel<<<(n + 255) / 256, b256, 0, stream>>>(dinv, n);
  count_deg_kernel<<<(unsigned)((E + 255) / 256), b256, 0, stream>>>(ei, dinv, E);
  dinv_kernel<<<(n + 255) / 256, b256, 0, stream>>>(dinv, n);
  prep_w_kernel<<<(HID * IN_C + OUT_C * HID + 255) / 256, b256, 0, stream>>>(W1, W2, w1t, w2t);

  const int gemmBlocks = (n + 63) / 64;
  gemm1_kernel<<<gemmBlocks, 128, 0, stream>>>(x, w1t, dinv, t1, agg1, n);

  const long long tot1 = E << 6;  // E * 64
  scatter_kernel<HID, 6><<<(unsigned)((tot1 + 255) / 256), b256, 0, stream>>>(ei, t1, dinv, agg1, E);

  const long long tot2 = (long long)n * HID;
  relu_bias_kernel<<<(unsigned)((tot2 + 255) / 256), b256, 0, stream>>>(agg1, b1, t1, tot2);

  gemm2_kernel<<<gemmBlocks, 128, 0, stream>>>(t1, w2t, dinv, b2, t2, out, n);

  const long long tot3 = E << 5;  // E * 32
  scatter_kernel<OUT_C, 5><<<(unsigned)((tot3 + 255) / 256), b256, 0, stream>>>(ei, t2, dinv, out, E);
}
